// Attention_82317343195415
// MI455X (gfx1250) — compile-verified, matrix operand order checked by reading, not run on hardware
//
#include <hip/hip_runtime.h>
#include <hip/hip_bf16.h>

// Problem constants (match reference)
#define Bc 4
#define Tt 4096
#define Ss 4096
#define Dd 256

typedef __attribute__((ext_vector_type(16))) __bf16 v16bf;
typedef __attribute__((ext_vector_type(8)))  __bf16 v8bf;
typedef __attribute__((ext_vector_type(8)))  float  v8f;

// ---------------------------------------------------------------------------
// Fragment loaders for V_WMMA_*_16X16X32_BF16 (wave32).
// 16-bit A-matrix layout (ISA 7.12.2): lane L -> row = L%16, half = L/16;
//   element e<8  : K = e      + 8*half
//   element e>=8 : K = e + 8  + 8*half
// => two contiguous 16B chunks at kbase+8*half and kbase+16+8*half.
// B-matrix mirrors this with N = L%16.
// ---------------------------------------------------------------------------
__device__ __forceinline__ v16bf frag_from_bf16_row(const __bf16* p, int half) {
    v8bf lo = *(const v8bf*)(p + 8 * half);
    v8bf hi = *(const v8bf*)(p + 16 + 8 * half);
    v16bf r;
#pragma unroll
    for (int i = 0; i < 8; ++i) { r[i] = lo[i]; r[8 + i] = hi[i]; }
    return r;
}

__device__ __forceinline__ v16bf frag_from_f32_row(const float* p, int half) {
    v8f lo = *(const v8f*)(p + 8 * half);
    v8f hi = *(const v8f*)(p + 16 + 8 * half);
    v16bf r;
#pragma unroll
    for (int i = 0; i < 8; ++i) { r[i] = (__bf16)lo[i]; r[8 + i] = (__bf16)hi[i]; }
    return r;
}

__device__ __forceinline__ v8f wmma_bf16(v16bf a, v16bf b, v8f c) {
    return __builtin_amdgcn_wmma_f32_16x16x32_bf16(false, a, false, b,
                                                   (short)0, c, false, false);
}

// --------------------------- K0: conversions -------------------------------
__global__ void k_cvt(const float* __restrict__ in, __bf16* __restrict__ out, int n) {
    int i = blockIdx.x * blockDim.x + threadIdx.x;
    int stride = gridDim.x * blockDim.x;
    for (; i < n; i += stride) out[i] = (__bf16)in[i];
}

// h_s [B][S][D] f32  ->  h_sT [B][D][S] bf16
__global__ void k_transpose_hs(const float* __restrict__ hs, __bf16* __restrict__ hsT) {
    int i = blockIdx.x * blockDim.x + threadIdx.x;
    const int total = Bc * Ss * Dd;
    int stride = gridDim.x * blockDim.x;
    for (; i < total; i += stride) {
        int d = i % Dd;
        int s = (i / Dd) % Ss;
        int b = i / (Dd * Ss);
        hsT[((size_t)b * Dd + d) * Ss + s] = (__bf16)hs[i];
    }
}

// --------------------------- K1: energy GEMM -------------------------------
// energy[b][s][n] = sum_k h_s[b][s][k] * W[n][k]
// One wave computes a 16(s) x 64(n) tile: 4 independent accumulators.
__global__ void __launch_bounds__(32)
k_energy(const float* __restrict__ hs, const __bf16* __restrict__ Wbf,
         __bf16* __restrict__ energy) {
    int tile = blockIdx.x;
    int n0 = (tile & 3) * 64;                   // Dd/64 = 4
    int st = (tile >> 2) & (Ss / 16 - 1);
    int b  = tile >> 10;
    int lane = threadIdx.x, l16 = lane & 15, half = lane >> 4;

    const float* arow = hs + ((size_t)b * Ss + st * 16 + l16) * Dd;
    const __bf16* brow[4];
#pragma unroll
    for (int j = 0; j < 4; ++j)
        brow[j] = Wbf + (size_t)(n0 + 16 * j + l16) * Dd;

    v8f c[4] = {};
#pragma unroll
    for (int kk = 0; kk < Dd; kk += 32) {
        v16bf a = frag_from_f32_row(arow + kk, half);
#pragma unroll
        for (int j = 0; j < 4; ++j)            // 4 independent WMMA chains
            c[j] = wmma_bf16(a, frag_from_bf16_row(brow[j] + kk, half), c[j]);
    }
    __bf16* orow = energy + ((size_t)b * Ss + st * 16) * Dd + n0;
#pragma unroll
    for (int j = 0; j < 4; ++j)
#pragma unroll
        for (int r = 0; r < 8; ++r)
            orow[(size_t)(r + 8 * half) * Dd + 16 * j + l16] = (__bf16)c[j][r];
}

// ----------------- K2: scores + masked softmax (fused) ---------------------
// One wave per (b, 16-row t-tile). Streams S in 64-column blocks
// (4 independent accumulators). Two passes:
//   pass 1: online rowwise (max, sumexp);  pass 2: recompute, write probs f32.
__global__ void __launch_bounds__(32)
k_scores_softmax(const __bf16* __restrict__ htb, const __bf16* __restrict__ energy,
                 const int* __restrict__ msk, float* __restrict__ probs) {
    int blk = blockIdx.x;
    int tt = blk & (Tt / 16 - 1);
    int b  = blk >> 8;                          // Tt/16 == 256
    int lane = threadIdx.x, l16 = lane & 15, half = lane >> 4;

    // Preload 16x256 h_t row-block as 8 A-fragments (lives in VGPRs)
    const __bf16* arow = htb + ((size_t)b * Tt + tt * 16 + l16) * Dd;
    v16bf aF[8];
#pragma unroll
    for (int kk = 0; kk < 8; ++kk) aF[kk] = frag_from_bf16_row(arow + 32 * kk, half);

    float m[8], l[8];
#pragma unroll
    for (int r = 0; r < 8; ++r) { m[r] = -1.0e30f; l[r] = 0.0f; }

    const int* mrow = msk + (size_t)b * Ss;
    const __bf16* ebase = energy + (size_t)b * Ss * Dd;

    // ---- pass 1: streaming logsumexp stats over 64-wide score blocks ----
    for (int s0 = 0; s0 < Ss; s0 += 64) {
        const __bf16* brow[4];
        int mk[4];
#pragma unroll
        for (int j = 0; j < 4; ++j) {
            brow[j] = ebase + (size_t)(s0 + 16 * j + l16) * Dd;
            mk[j] = mrow[s0 + 16 * j + l16];
        }
        __builtin_prefetch((const void*)(brow[3] + 16 * Dd), 0, 1);
        v8f c[4] = {};
#pragma unroll
        for (int kk = 0; kk < 8; ++kk)
#pragma unroll
            for (int j = 0; j < 4; ++j)
                c[j] = wmma_bf16(aF[kk], frag_from_bf16_row(brow[j] + 32 * kk, half), c[j]);

#pragma unroll
        for (int r = 0; r < 8; ++r) {
            // mask: score*m - 1e11*(1-m), then clamp(-1e10) == m ? score : -1e10
            float x0 = mk[0] ? c[0][r] : -1.0e10f;
            float x1 = mk[1] ? c[1][r] : -1.0e10f;
            float x2 = mk[2] ? c[2][r] : -1.0e10f;
            float x3 = mk[3] ? c[3][r] : -1.0e10f;
            float tm = fmaxf(fmaxf(x0, x1), fmaxf(x2, x3));
#pragma unroll
            for (int sh = 1; sh < 16; sh <<= 1)      // rowwise max within half
                tm = fmaxf(tm, __shfl_xor(tm, sh, 32));
            float nm = fmaxf(m[r], tm);
            float e = __expf(x0 - nm) + __expf(x1 - nm) +
                      __expf(x2 - nm) + __expf(x3 - nm);
#pragma unroll
            for (int sh = 1; sh < 16; sh <<= 1)      // rowwise sum within half
                e += __shfl_xor(e, sh, 32);
            l[r] = l[r] * __expf(m[r] - nm) + e;
            m[r] = nm;
        }
    }
    float rl[8];
#pragma unroll
    for (int r = 0; r < 8; ++r) rl[r] = 1.0f / l[r];

    // ---- pass 2: recompute scores, write normalized probs ----
    for (int s0 = 0; s0 < Ss; s0 += 64) {
        const __bf16* brow[4];
        int mk[4];
#pragma unroll
        for (int j = 0; j < 4; ++j) {
            brow[j] = ebase + (size_t)(s0 + 16 * j + l16) * Dd;
            mk[j] = mrow[s0 + 16 * j + l16];
        }
        __builtin_prefetch((const void*)(brow[3] + 16 * Dd), 0, 1);
        v8f c[4] = {};
#pragma unroll
        for (int kk = 0; kk < 8; ++kk)
#pragma unroll
            for (int j = 0; j < 4; ++j)
                c[j] = wmma_bf16(aF[kk], frag_from_bf16_row(brow[j] + 32 * kk, half), c[j]);

#pragma unroll
        for (int j = 0; j < 4; ++j) {
            float* prow = probs + ((size_t)(b * Tt + tt * 16)) * Ss + s0 + 16 * j + l16;
#pragma unroll
            for (int r = 0; r < 8; ++r) {
                float x = mk[j] ? c[j][r] : -1.0e10f;
                prow[(size_t)(r + 8 * half) * Ss] = __expf(x - m[r]) * rl[r];
            }
        }
    }
}

// --------------------------- K3: context GEMM ------------------------------
// context[b][t][d] = sum_s probs[b][t][s] * h_s[b][s][d]
// One wave computes a 16(t) x 64(d) tile: 4 accumulator chains live across S.
__global__ void __launch_bounds__(32)
k_context(const float* __restrict__ probs, const __bf16* __restrict__ hsT,
          float* __restrict__ ctx) {
    int tile = blockIdx.x;
    int d0 = (tile & 3) * 64;                   // Dd/64 = 4
    int tt = (tile >> 2) & (Tt / 16 - 1);
    int b  = tile >> 10;
    int lane = threadIdx.x, l16 = lane & 15, half = lane >> 4;

    const float* arow = probs + ((size_t)(b * Tt + tt * 16 + l16)) * Ss;
    const __bf16* brow[4];
#pragma unroll
    for (int j = 0; j < 4; ++j)
        brow[j] = hsT + ((size_t)b * Dd + d0 + 16 * j + l16) * Ss;

    v8f c[4] = {};
    for (int ss = 0; ss < Ss; ss += 32) {
        __builtin_prefetch((const void*)(arow + ss + 256), 0, 1);
        v16bf a = frag_from_f32_row(arow + ss, half);
#pragma unroll
        for (int j = 0; j < 4; ++j)
            c[j] = wmma_bf16(a, frag_from_bf16_row(brow[j] + ss, half), c[j]);
    }
    float* orow = ctx + ((size_t)(b * Tt + tt * 16)) * Dd + d0;
#pragma unroll
    for (int j = 0; j < 4; ++j)
#pragma unroll
        for (int r = 0; r < 8; ++r)
            orow[(size_t)(r + 8 * half) * Dd + 16 * j + l16] = c[j][r];
}

// ---------------------------------------------------------------------------
extern "C" void kernel_launch(void* const* d_in, const int* in_sizes, int n_in,
                              void* d_out, int out_size, void* d_ws, size_t ws_size,
                              hipStream_t stream) {
    const float* h_t = (const float*)d_in[0];   // (B,T,256) f32
    const float* h_s = (const float*)d_in[1];   // (B,S,256) f32
    const int*   m_s = (const int*)d_in[2];     // (B,S)     i32
    const float* Win = (const float*)d_in[3];   // (256,256) f32

    // workspace layout (bf16): h_t | W | energy | h_sT  (~25.3 MB total)
    __bf16* htb    = (__bf16*)d_ws;
    __bf16* Wbf    = htb    + (size_t)Bc * Tt * Dd;
    __bf16* energy = Wbf    + (size_t)Dd * Dd;
    __bf16* hsT    = energy + (size_t)Bc * Ss * Dd;

    float* ctx   = (float*)d_out;               // (B,T,256) f32
    float* probs = ctx + (size_t)Bc * Tt * Dd;  // (B,T,S)   f32

    k_cvt<<<2048, 256, 0, stream>>>(h_t, htb, Bc * Tt * Dd);
    k_cvt<<<64, 256, 0, stream>>>(Win, Wbf, Dd * Dd);
    k_transpose_hs<<<2048, 256, 0, stream>>>(h_s, hsT);

    k_energy<<<Bc * (Ss / 16) * (Dd / 64), 32, 0, stream>>>(h_s, Wbf, energy);
    k_scores_softmax<<<Bc * (Tt / 16), 32, 0, stream>>>(htb, energy, m_s, probs);
    k_context<<<Bc * (Tt / 16) * (Dd / 64), 32, 0, stream>>>(probs, hsT, ctx);
}
